// Res_DCN_31628139167828
// MI455X (gfx1250) — compile-verified, matrix operand order checked by reading, not run on hardware
//
#include <hip/hip_runtime.h>
#include <math.h>

// Problem constants (B, C, H, W fixed by the reference)
#define BATCH  4
#define CH     64
#define HH     128
#define WW     128
#define K2     9
#define CK     576   // CH * K2
#define OC_OFF 27    // offset-conv output channels
#define OC_PAD 32    // padded to 2 WMMA M-tiles

typedef __attribute__((ext_vector_type(16))) _Float16 v16h;
typedef __attribute__((ext_vector_type(8)))  float    v8f;
typedef __attribute__((ext_vector_type(2)))  float    f2v;

union ABFrag { v16h v; uint4 u[2]; };

// per-wave scheduling fence: per-wave LDS ops are in-order (DScnt), we only
// need to stop the compiler from reordering ds stores/loads within the wave.
#define WAVE_FENCE() __builtin_amdgcn_wave_barrier()

// 8-byte gather that tolerates 4-byte alignment (backend may use b64@align4)
__device__ __forceinline__ f2v load_f2(const float* p) {
  f2v r;
  __builtin_memcpy(&r, p, 8);
  return r;
}

// ---------------------------------------------------------------------------
// Weight conversion kernels (fp32 -> f16). Main weights [64][576] are already
// flat in (o, c*9+k) order. Offset weights are padded 27 -> 32 rows with zeros.
// ---------------------------------------------------------------------------
__global__ void convert_w_kernel(const float* __restrict__ w,
                                 _Float16* __restrict__ wh) {
  int i = blockIdx.x * blockDim.x + threadIdx.x;
  if (i < CH * CK) wh[i] = (_Float16)w[i];
}

__global__ void convert_woff_kernel(const float* __restrict__ w,
                                    _Float16* __restrict__ wh) {
  int i = blockIdx.x * blockDim.x + threadIdx.x;
  if (i >= OC_PAD * CK) return;
  int o = i / CK;
  int ck = i - o * CK;
  wh[i] = (o < OC_OFF) ? (_Float16)w[o * CK + ck] : (_Float16)0.f;
}

// ---------------------------------------------------------------------------
// Offset conv as implicit-im2col WMMA GEMM: om[32pad, px] = Woff[32,576] x P.
// One block per (b, row); wave i owns pixels [16i, 16i+16). Per-wave S tile
// [16 px][stride 56 halves] feeds aligned b128 B-fragment ds loads.
// ---------------------------------------------------------------------------
__global__ void __launch_bounds__(256)
offset_wmma_kernel(const float* __restrict__ xin,
                   const _Float16* __restrict__ woh,   // [32][576] f16, padded
                   const float* __restrict__ b_off,
                   float* __restrict__ om) {
  __shared__ __align__(16) _Float16 S[8 * 16 * 56];   // 14 KB

  const int tid  = threadIdx.x;
  const int wave = tid >> 5;
  const int lane = tid & 31;
  const int col  = lane & 15;
  const int half = lane >> 4;
  const int b    = blockIdx.x / HH;
  const int h    = blockIdx.x % HH;
  const int w0   = wave * 16;

  v8f acc[2];
#pragma unroll
  for (int mt = 0; mt < 2; ++mt)
#pragma unroll
    for (int r = 0; r < 8; ++r) acc[mt][r] = 0.f;

  const float* xb = xin + (size_t)b * CH * HH * WW;
  _Float16*    Sw = S + wave * (16 * 56);

  for (int ks = 0; ks < 18; ++ks) {
    // implicit im2col: S[r][col] = x[c, h-1+ky, w-1+kx] (0 outside)
#pragma unroll 4
    for (int i = 0; i < 16; ++i) {
      int r  = i * 2 + half;
      int ck = ks * 32 + r;
      int c  = ck / 9;
      int k  = ck - c * 9;
      int ky = k / 3;
      int kx = k - ky * 3;
      int y  = h - 1 + ky;
      int xw = w0 + col - 1 + kx;
      bool inb = ((unsigned)y < HH) && ((unsigned)xw < WW);
      int yc  = min(max(y, 0), HH - 1);
      int xcc = min(max(xw, 0), WW - 1);
      float v = xb[(size_t)c * (HH * WW) + yc * WW + xcc];
      Sw[col * 56 + r] = (_Float16)(inb ? v : 0.f);
    }
    WAVE_FENCE();

    const uint4* bp = (const uint4*)(Sw + col * 56);
    ABFrag bf;
    bf.u[0] = bp[half];
    bf.u[1] = bp[2 + half];
#pragma unroll
    for (int mt = 0; mt < 2; ++mt) {
      const _Float16* wr = woh + (size_t)(mt * 16 + col) * CK + ks * 32 + half * 8;
      ABFrag af;
      af.u[0] = ((const uint4*)wr)[0];
      af.u[1] = ((const uint4*)wr)[2];
      acc[mt] = __builtin_amdgcn_wmma_f32_16x16x32_f16(
          false, af.v, false, bf.v, (short)0, acc[mt], false, false);
    }
    WAVE_FENCE();
  }

#pragma unroll
  for (int mt = 0; mt < 2; ++mt) {
#pragma unroll
    for (int r = 0; r < 8; ++r) {
      int o = mt * 16 + half * 8 + r;
      if (o < OC_OFF) {
        om[((size_t)(b * 27 + o) * HH + h) * WW + w0 + col] = acc[mt][r] + b_off[o];
      }
    }
  }
}

// ---------------------------------------------------------------------------
// Fused bilinear sampling + mask + WMMA contraction + epilogue.
// MODE 0: ReLU epilogue (layer 1). MODE 1: residual-add epilogue (layer 2).
// ---------------------------------------------------------------------------
template <int MODE>
__global__ void __launch_bounds__(256)
dcn_wmma_kernel(const float* __restrict__ xin,
                const float* __restrict__ om,
                const _Float16* __restrict__ wh,   // [64][576] f16
                const float* __restrict__ bias,
                float* __restrict__ out,
                const float* __restrict__ res) {
  // Per-wave bilinear params for 144 (k, pixel) pairs (shared across 64 ch).
  // Pair-aligned x-weights (wl, wr) let each row pair be one 8-byte gather:
  // any clamped/OOB corner already carries weight 0.
  __shared__ float4 pw[8 * 144];                      // (wy0m, wy1m, wl, wr)
  __shared__ int2   pi2[8 * 144];                     // pair base offsets rows 0/1
  __shared__ __align__(16) _Float16 S[8 * 16 * 56];   // sampled tile per wave

  const int tid  = threadIdx.x;
  const int wave = tid >> 5;
  const int lane = tid & 31;
  const int col  = lane & 15;
  const int half = lane >> 4;
  const int b    = blockIdx.x / HH;
  const int h    = blockIdx.x % HH;
  const int w0   = wave * 16;

  // ---- phase A: per-(k,pixel) params
  const float* omb = om + (size_t)b * 27 * HH * WW + (size_t)h * WW;
  for (int p = lane; p < 144; p += 32) {
    int k  = p >> 4;
    int px = p & 15;
    int wp = w0 + px;
    float dy = omb[(size_t)(2 * k) * (HH * WW) + wp];
    float dx = omb[(size_t)(2 * k + 1) * (HH * WW) + wp];
    float mr = omb[(size_t)(18 + k) * (HH * WW) + wp];
    float mask = 1.0f / (1.0f + __expf(-mr));
    float py  = (float)(h - 1 + (k / 3)) + dy;
    float pxf = (float)(wp - 1 + (k % 3)) + dx;
    float fy0 = floorf(py), fx0 = floorf(pxf);
    float wy1 = py - fy0,  wx1 = pxf - fx0;
    float wy0 = 1.f - wy1, wx0 = 1.f - wx1;
    int iy0 = (int)fy0, ix0 = (int)fx0;
    int iy1 = iy0 + 1,  ix1 = ix0 + 1;
    float vy0 = (iy0 >= 0 && iy0 < HH) ? 1.f : 0.f;
    float vy1 = (iy1 >= 0 && iy1 < HH) ? 1.f : 0.f;
    float vx0 = (ix0 >= 0 && ix0 < WW) ? 1.f : 0.f;
    float vx1 = (ix1 >= 0 && ix1 < WW) ? 1.f : 0.f;
    float wx0v = wx0 * vx0, wx1v = wx1 * vx1;
    // pair-aligned x-weights against base = clamp(ix0, 0, W-2)
    int xbase = min(max(ix0, 0), WW - 2);
    float wl = (ix0 == xbase     ? wx0v : 0.f) + (ix1 == xbase     ? wx1v : 0.f);
    float wr = (ix0 == xbase + 1 ? wx0v : 0.f) + (ix1 == xbase + 1 ? wx1v : 0.f);
    int cy0 = min(max(iy0, 0), HH - 1);
    int cy1 = min(max(iy1, 0), HH - 1);
    pw[wave * 144 + p]  = make_float4(mask * wy0 * vy0, mask * wy1 * vy1, wl, wr);
    pi2[wave * 144 + p] = make_int2(cy0 * WW + xbase, cy1 * WW + xbase);
  }
  WAVE_FENCE();

  v8f acc[4];
#pragma unroll
  for (int mt = 0; mt < 4; ++mt)
#pragma unroll
    for (int r = 0; r < 8; ++r) acc[mt][r] = 0.f;

  const float* xb   = xin + (size_t)b * CH * HH * WW;
  _Float16*    Sw   = S   + wave * (16 * 56);
  const float4* pwp = pw  + wave * 144;
  const int2*   pip = pi2 + wave * 144;

  for (int ks = 0; ks < 18; ++ks) {
    // ---- phase B: sample 32 (c,k) rows x 16 px -> f16 tile (2x b64 gathers)
#pragma unroll 4
    for (int i = 0; i < 16; ++i) {
      int r  = i * 2 + half;
      int ck = ks * 32 + r;
      int c  = ck / 9;
      int k  = ck - c * 9;
      float4 wv = pwp[k * 16 + col];
      int2   iv = pip[k * 16 + col];
      const float* xc = xb + (size_t)c * (HH * WW);
      f2v ra = load_f2(xc + iv.x);
      f2v rb = load_f2(xc + iv.y);
      float s = wv.x * (wv.z * ra.x + wv.w * ra.y) +
                wv.y * (wv.z * rb.x + wv.w * rb.y);
      Sw[col * 56 + r] = (_Float16)s;
    }
    WAVE_FENCE();

    // ---- B fragment: two aligned b128 LDS loads per lane
    const uint4* bp = (const uint4*)(Sw + col * 56);
    ABFrag bf;
    bf.u[0] = bp[half];
    bf.u[1] = bp[2 + half];
#pragma unroll
    for (int mt = 0; mt < 4; ++mt) {
      const _Float16* wr = wh + (size_t)(mt * 16 + col) * CK + ks * 32 + half * 8;
      ABFrag af;
      af.u[0] = ((const uint4*)wr)[0];
      af.u[1] = ((const uint4*)wr)[2];   // +16 halves = +32 B
      acc[mt] = __builtin_amdgcn_wmma_f32_16x16x32_f16(
          false, af.v, false, bf.v, (short)0, acc[mt], false, false);
    }
    WAVE_FENCE();
  }

  // ---- epilogue: C/D layout -> M = half*8 + vgpr, N = col
#pragma unroll
  for (int mt = 0; mt < 4; ++mt) {
#pragma unroll
    for (int r = 0; r < 8; ++r) {
      int o = mt * 16 + half * 8 + r;
      float v = acc[mt][r] + bias[o];
      size_t oi = (((size_t)b * CH + o) * HH + h) * WW + w0 + col;
      if (MODE == 0) v = fmaxf(v, 0.f);
      else           v += res[oi];
      out[oi] = v;
    }
  }
}

// ---------------------------------------------------------------------------
// Host launch: x -> om1 -> dcn1(relu) -> h -> om2 -> dcn2(+x) -> out
// ---------------------------------------------------------------------------
extern "C" void kernel_launch(void* const* d_in, const int* in_sizes, int n_in,
                              void* d_out, int out_size, void* d_ws, size_t ws_size,
                              hipStream_t stream) {
  const float* x      = (const float*)d_in[0];
  const float* w_off1 = (const float*)d_in[1];
  const float* b_off1 = (const float*)d_in[2];
  const float* w1     = (const float*)d_in[3];
  const float* b1     = (const float*)d_in[4];
  const float* w_off2 = (const float*)d_in[5];
  const float* b_off2 = (const float*)d_in[6];
  const float* w2     = (const float*)d_in[7];
  const float* b2     = (const float*)d_in[8];
  float* out = (float*)d_out;

  char* ws = (char*)d_ws;
  const size_t WH_BYTES  = (size_t)CH * CK * sizeof(_Float16);            //  73,728
  const size_t WOH_BYTES = (size_t)OC_PAD * CK * sizeof(_Float16);        //  36,864
  const size_t OM_BYTES  = (size_t)BATCH * 27 * HH * WW * sizeof(float);  // 7.08 MB
  _Float16* wh1  = (_Float16*)ws;
  _Float16* wh2  = (_Float16*)(ws + WH_BYTES);
  _Float16* woh1 = (_Float16*)(ws + 2 * WH_BYTES);
  _Float16* woh2 = (_Float16*)(ws + 2 * WH_BYTES + WOH_BYTES);
  float*    omb  = (float*)(ws + 2 * WH_BYTES + 2 * WOH_BYTES);
  float*    hb   = (float*)(ws + 2 * WH_BYTES + 2 * WOH_BYTES + OM_BYTES);

  const int nrow = BATCH * HH;   // 512 blocks: one per (batch, row)

  convert_w_kernel<<<(CH * CK + 255) / 256, 256, 0, stream>>>(w1, wh1);
  convert_w_kernel<<<(CH * CK + 255) / 256, 256, 0, stream>>>(w2, wh2);
  convert_woff_kernel<<<(OC_PAD * CK + 255) / 256, 256, 0, stream>>>(w_off1, woh1);
  convert_woff_kernel<<<(OC_PAD * CK + 255) / 256, 256, 0, stream>>>(w_off2, woh2);

  // layer 1
  offset_wmma_kernel<<<nrow, 256, 0, stream>>>(x, woh1, b_off1, omb);
  dcn_wmma_kernel<0><<<nrow, 256, 0, stream>>>(x, omb, wh1, b1, hb, nullptr);

  // layer 2 + residual
  offset_wmma_kernel<<<nrow, 256, 0, stream>>>(hb, woh2, b_off2, omb);
  dcn_wmma_kernel<1><<<nrow, 256, 0, stream>>>(hb, omb, wh2, b2, out, x);
}
